// Net_30648886624676
// MI455X (gfx1250) — compile-verified, matrix operand order checked by reading, not run on hardware
//
#include <hip/hip_runtime.h>
#include <hip/hip_bf16.h>
#include <math.h>

#define BB 32
#define NN 2048
#define EE 32768
#define HH 128

typedef __attribute__((ext_vector_type(16))) __bf16 v16bf;
typedef __attribute__((ext_vector_type(8)))  __bf16 v8bf;
typedef __attribute__((ext_vector_type(8)))  float  v8f;

static __device__ __forceinline__ __bf16 f2bf(float f) {
  union { float f; unsigned u; } x; x.f = f;
  unsigned r = (x.u + 0x7FFFu + ((x.u >> 16) & 1u)) >> 16;  // RNE truncate to bf16
  unsigned short h = (unsigned short)r;
  return __builtin_bit_cast(__bf16, h);
}

// ---------------------------------------------------------------------------
// Prep: cast node features to bf16 (same row-major layout).
__global__ void cast_bf16(const float* __restrict__ in, __bf16* out, int total) {
  int i = blockIdx.x * 256 + threadIdx.x;
  if (i < total) out[i] = f2bf(in[i]);
}

// Prep: W[128x128] f32 row-major -> Wt bf16 col-major (Wt[c*128+k] = W[k*128+c]).
__global__ void cast_w_transpose(const float* __restrict__ W, __bf16* Wt) {
  int i = blockIdx.x * 256 + threadIdx.x;   // 16384 elements
  int k = i & 127, c = i >> 7;
  Wt[i] = f2bf(W[k * HH + c]);
}

// ---------------------------------------------------------------------------
// C[n x 128] = A[n x 128] @ W[128 x 128] per graph; A,Wt pre-converted bf16.
// One wave = one 16x16 output tile; 8 waves/block cover the 8 column tiles.
// CDNA5 16-bit A-matrix lane layout (ISA 7.12.2):
//   lanes 0-15  : row M=lane,    VGPR j/2 holds K = {0..7, 16..23}
//   lanes 16-31 : row M=lane-16, K = {8..15, 24..31}
// => per lane the 16 operand values are two contiguous 8-element (16B) runs:
//    K = kb*32 + half*8 + [0..7]  and  K = kb*32 + 16 + half*8 + [0..7]
// B mirrored with N in place of M (Wt is col-major so runs are contiguous too).
// C layout: VGPR r -> M = r + 8*half, N = lane%16.
// ---------------------------------------------------------------------------
__global__ void gemm_xw_wmma(const __bf16* __restrict__ A, long long astride,
                             const __bf16* __restrict__ Wt,
                             float* __restrict__ C, long long cstride) {
  int wave = threadIdx.x >> 5;
  int lane = threadIdx.x & 31;
  int half = lane >> 4;
  int l16  = lane & 15;
  int b    = blockIdx.y;
  int row0 = blockIdx.x * 16;
  int col0 = wave * 16;
  const __bf16* Ab = A + (long long)b * astride + (long long)(row0 + l16) * HH;
  const __bf16* Wb = Wt + (long long)(col0 + l16) * HH;
  float*        Cb = C + (long long)b * cstride + (long long)row0 * HH;

  v8f acc = {};
#pragma unroll
  for (int kb = 0; kb < 4; ++kb) {
    int k0 = kb * 32 + half * 8;
    v8bf a0 = *(const v8bf*)(Ab + k0);
    v8bf a1 = *(const v8bf*)(Ab + k0 + 16);
    v8bf b0 = *(const v8bf*)(Wb + k0);
    v8bf b1 = *(const v8bf*)(Wb + k0 + 16);
    v16bf av, bv;
#pragma unroll
    for (int j = 0; j < 8; ++j) {
      av[j] = a0[j]; av[j + 8] = a1[j];
      bv[j] = b0[j]; bv[j + 8] = b1[j];
    }
    acc = __builtin_amdgcn_wmma_f32_16x16x32_bf16(false, av, false, bv,
                                                  (short)0, acc, false, false);
  }
#pragma unroll
  for (int r = 0; r < 8; ++r)
    Cb[(r + half * 8) * HH + col0 + l16] = acc[r];
}

// ---------------------------------------------------------------------------
__global__ void copy_edges(const int* __restrict__ ei, const float* __restrict__ ea,
                           int* esrc, int* edst, float* ew, int total) {
  int i = blockIdx.x * 256 + threadIdx.x;
  if (i >= total) return;
  esrc[i] = ei[(long long)i * 2];
  edst[i] = ei[(long long)i * 2 + 1];
  ew[i]   = ea[i];
}

__global__ void fill_f32(float* p, int total, float v) {
  int i = blockIdx.x * 256 + threadIdx.x;
  if (i < total) p[i] = v;
}

__global__ void deg_acc(const int* __restrict__ edst, const float* __restrict__ ew,
                        float* deg, int n, int total) {
  int i = blockIdx.x * 256 + threadIdx.x;
  if (i >= total) return;
  int b = i >> 15;                       // / EE (32768)
  atomicAdd(&deg[b * n + edst[i]], ew[i]);
}

__global__ void deg_to_dinv(float* d, int total) {
  int i = blockIdx.x * 256 + threadIdx.x;
  if (i < total) d[i] = rsqrtf(d[i]);
}

// hg = hl * dinv^2 (self-loop term; also serves as zero-init of the accumulator)
__global__ void self_init(const float* __restrict__ hl, const float* __restrict__ dinv,
                          float* hg, int total) {
  int i = blockIdx.x * 256 + threadIdx.x;
  if (i >= total) return;
  float dv = dinv[i >> 7];               // / HH
  hg[i] = hl[i] * dv * dv;
}

// one block per edge, 128 threads = features: hg[dst] += norm * hl[src]
__global__ void edge_agg(const int* __restrict__ esrc, const int* __restrict__ edst,
                         const float* __restrict__ ew, const float* __restrict__ dinv,
                         const float* __restrict__ hl, float* hg, int n) {
  int b = blockIdx.y;
  long long e = (long long)b * EE + blockIdx.x;
  int s = esrc[e], d = edst[e];
  float w = ew[e];
  float norm = dinv[b * n + s] * w * dinv[b * n + d];
  if (norm != 0.f) {
    int f = threadIdx.x;
    atomicAdd(&hg[((long long)(b * n + d)) * HH + f],
              norm * hl[((long long)(b * n + s)) * HH + f]);
  }
}

__global__ void add_bias_relu(float* hg, const float* __restrict__ bias, int total) {
  int i = blockIdx.x * 256 + threadIdx.x;
  if (i >= total) return;
  hg[i] = fmaxf(hg[i] + bias[i & 127], 0.f);
}

__global__ void gemv_score(const float* __restrict__ hg, const float* __restrict__ Ws,
                           float* sl, int total) {
  int i = blockIdx.x * 256 + threadIdx.x;
  if (i >= total) return;
  float a = 0.f;
#pragma unroll
  for (int j = 0; j < HH; ++j) a += hg[(long long)i * HH + j] * Ws[j];
  sl[i] = a;
}

__global__ void score_self(const float* __restrict__ sl, const float* __restrict__ dinv,
                           float* sc, int total) {
  int i = blockIdx.x * 256 + threadIdx.x;
  if (i >= total) return;
  float dv = dinv[i];
  sc[i] = sl[i] * dv * dv;
}

__global__ void score_edge(const int* __restrict__ esrc, const int* __restrict__ edst,
                           const float* __restrict__ ew, const float* __restrict__ dinv,
                           const float* __restrict__ sl, float* sc, int n, int total) {
  int i = blockIdx.x * 256 + threadIdx.x;
  if (i >= total) return;
  int b = i >> 15, base = b * n;
  int s = esrc[i], d = edst[i];
  float norm = dinv[base + s] * ew[i] * dinv[base + d];
  if (norm != 0.f) atomicAdd(&sc[base + d], norm * sl[base + s]);
}

// Per-graph bitonic sort (descending, ties -> lower index) + top-k bookkeeping.
__global__ __launch_bounds__(1024) void sort_pool(
    const float* __restrict__ sc, const float* __restrict__ bs, int n, int k,
    float* scs, int* perm, int* kept, int* remap) {
  __shared__ float skey[NN];
  __shared__ int   sidx[NN];
  int b = blockIdx.x, t = threadIdx.x;
  float bsv = bs[0];
  for (int i = t; i < n; i += 1024) { skey[i] = sc[b * n + i] + bsv; sidx[i] = i; }
  __syncthreads();
  for (int size = 2; size <= n; size <<= 1) {
    for (int stride = size >> 1; stride > 0; stride >>= 1) {
      int tpairs = n >> 1;
      if (t < tpairs) {
        int i = (t / stride) * (stride << 1) + (t % stride);
        int j = i + stride;
        bool descSeg = ((i & size) == 0);
        float ki = skey[i], kj = skey[j];
        int ii = sidx[i], ij = sidx[j];
        bool bef = (ki > kj) || (ki == kj && ii < ij);   // i before j in DESC order
        bool sw = descSeg ? (!bef) : bef;
        if (sw) { skey[i] = kj; skey[j] = ki; sidx[i] = ij; sidx[j] = ii; }
      }
      __syncthreads();
    }
  }
  for (int i = t; i < n; i += 1024) {
    int node = sidx[i];
    int inK = (i < k) ? 1 : 0;
    if (inK) { scs[b * n + i] = skey[i]; perm[b * n + i] = node; }
    kept[b * n + node]  = inK;
    remap[b * n + node] = inK ? i : 0;
  }
}

__global__ void gather_gate(const float* __restrict__ hg, const float* __restrict__ scs,
                            const int* __restrict__ perm, float* hout, int n, int k) {
  int b = blockIdx.y, i = blockIdx.x, f = threadIdx.x;
  int node = perm[b * n + i];
  float g = tanhf(scs[b * n + i]);
  hout[((long long)(b * k + i)) * HH + f] = hg[((long long)(b * n + node)) * HH + f] * g;
}

__global__ void readout_acc(const float* __restrict__ h, int k, float* r) {
  int b = blockIdx.x, f = threadIdx.x;
  float mx = -INFINITY, sm = 0.f;
  for (int i = 0; i < k; ++i) {
    float v = h[((long long)(b * k + i)) * HH + f];
    mx = fmaxf(mx, v);
    sm += v;
  }
  r[b * 256 + f]       += mx;
  r[b * 256 + 128 + f] += sm / (float)k;
}

__global__ void edge_filter(int* esrc, int* edst, float* ew,
                            const int* __restrict__ kept, const int* __restrict__ remap,
                            int n, int total) {
  int i = blockIdx.x * 256 + threadIdx.x;
  if (i >= total) return;
  int b = i >> 15, base = b * n;
  int s = esrc[i], d = edst[i];
  bool v = kept[base + s] && kept[base + d];
  esrc[i] = v ? remap[base + s] : 0;
  edst[i] = v ? remap[base + d] : 0;
  ew[i]   = v ? ew[i] : 0.f;
}

__global__ void mlp_head(const float* __restrict__ r,
                         const float* __restrict__ L1w, const float* __restrict__ L1b,
                         const float* __restrict__ L2w, const float* __restrict__ L2b,
                         const float* __restrict__ L3w, const float* __restrict__ L3b,
                         float* out) {
  __shared__ float h1[128], h2[64], lg[2];
  int b = blockIdx.x, t = threadIdx.x;
  float a = 0.f;
  for (int j = 0; j < 256; ++j) a += r[b * 256 + j] * L1w[j * 128 + t];
  h1[t] = fmaxf(a + L1b[t], 0.f);
  __syncthreads();
  if (t < 64) {
    float a2 = 0.f;
    for (int j = 0; j < 128; ++j) a2 += h1[j] * L2w[j * 64 + t];
    h2[t] = fmaxf(a2 + L2b[t], 0.f);
  }
  __syncthreads();
  if (t < 2) {
    float a3 = 0.f;
    for (int j = 0; j < 64; ++j) a3 += h2[j] * L3w[j * 2 + t];
    lg[t] = a3 + L3b[t];
  }
  __syncthreads();
  if (t < 2) {
    float m = fmaxf(lg[0], lg[1]);
    float lse = m + logf(expf(lg[0] - m) + expf(lg[1] - m));
    out[b * 2 + t] = lg[t] - lse;
  }
}

// ---------------------------------------------------------------------------
extern "C" void kernel_launch(void* const* d_in, const int* in_sizes, int n_in,
                              void* d_out, int out_size, void* d_ws, size_t ws_size,
                              hipStream_t stream) {
  (void)in_sizes; (void)n_in; (void)out_size; (void)ws_size;

  size_t off = 0;
  auto carve = [&](size_t bytes) -> void* {
    void* q = (char*)d_ws + off;
    off += (bytes + 255) & ~(size_t)255;
    return q;
  };
  float*  hbuf  = (float*)carve((size_t)BB * NN * HH * 4);  // pooled features (next-stage input)
  float*  hlin  = (float*)carve((size_t)BB * NN * HH * 4);  // h @ W
  float*  hgcn  = (float*)carve((size_t)BB * NN * HH * 4);  // gcn output
  __bf16* abf   = (__bf16*)carve((size_t)BB * NN * HH * 2); // bf16 copy of GEMM input
  __bf16* wtbf  = (__bf16*)carve((size_t)HH * HH * 2);      // bf16 transposed weight
  float*  slin  = (float*)carve((size_t)BB * NN * 4);
  float*  ssc   = (float*)carve((size_t)BB * NN * 4);
  float*  dinv  = (float*)carve((size_t)BB * NN * 4);
  float*  scs   = (float*)carve((size_t)BB * NN * 4);
  int*    perm  = (int*)carve((size_t)BB * NN * 4);
  int*    kept  = (int*)carve((size_t)BB * NN * 4);
  int*    remap = (int*)carve((size_t)BB * NN * 4);
  int*    esrc  = (int*)carve((size_t)BB * EE * 4);
  int*    edst  = (int*)carve((size_t)BB * EE * 4);
  float*  ew    = (float*)carve((size_t)BB * EE * 4);
  float*  rbuf  = (float*)carve((size_t)BB * 256 * 4);

  const float* x  = (const float*)d_in[0];
  const int*   ei = (const int*)d_in[1];
  const float* ea = (const float*)d_in[2];
  const float* Wl[3]  = {(const float*)d_in[3],  (const float*)d_in[7],  (const float*)d_in[11]};
  const float* bl[3]  = {(const float*)d_in[4],  (const float*)d_in[8],  (const float*)d_in[12]};
  const float* Wsl[3] = {(const float*)d_in[5],  (const float*)d_in[9],  (const float*)d_in[13]};
  const float* bsl[3] = {(const float*)d_in[6],  (const float*)d_in[10], (const float*)d_in[14]};

  const int totalE = BB * EE;
  copy_edges<<<(totalE + 255) / 256, 256, 0, stream>>>(ei, ea, esrc, edst, ew, totalE);
  fill_f32<<<(BB * 256 + 255) / 256, 256, 0, stream>>>(rbuf, BB * 256, 0.f);

  int n = NN;
  const float* hin = x;
  for (int s = 0; s < 3; ++s) {
    int k  = n >> 1;
    int tn = BB * n;
    int tf = BB * n * HH;
    // 1) dense: hlin = hin @ W  (bf16 staging + WMMA)
    cast_bf16<<<(tf + 255) / 256, 256, 0, stream>>>(hin, abf, tf);
    cast_w_transpose<<<(HH * HH + 255) / 256, 256, 0, stream>>>(Wl[s], wtbf);
    gemm_xw_wmma<<<dim3(n / 16, BB), 256, 0, stream>>>(abf, (long long)n * HH, wtbf,
                                                       hlin, (long long)n * HH);
    // 2) degree -> dinv
    fill_f32<<<(tn + 255) / 256, 256, 0, stream>>>(dinv, tn, 1.0f);
    deg_acc<<<(totalE + 255) / 256, 256, 0, stream>>>(edst, ew, dinv, n, totalE);
    deg_to_dinv<<<(tn + 255) / 256, 256, 0, stream>>>(dinv, tn);
    // 3) aggregate + bias + relu
    self_init<<<(tf + 255) / 256, 256, 0, stream>>>(hlin, dinv, hgcn, tf);
    edge_agg<<<dim3(EE, BB), HH, 0, stream>>>(esrc, edst, ew, dinv, hlin, hgcn, n);
    add_bias_relu<<<(tf + 255) / 256, 256, 0, stream>>>(hgcn, bl[s], tf);
    // 4) score GCN (GEMV + same-normalized aggregation; bias folded into sort)
    gemv_score<<<(tn + 255) / 256, 256, 0, stream>>>(hgcn, Wsl[s], slin, tn);
    score_self<<<(tn + 255) / 256, 256, 0, stream>>>(slin, dinv, ssc, tn);
    score_edge<<<(totalE + 255) / 256, 256, 0, stream>>>(esrc, edst, ew, dinv, slin, ssc, n, totalE);
    // 5) per-graph top-k via bitonic sort in LDS
    sort_pool<<<BB, 1024, 0, stream>>>(ssc, bsl[s], n, k, scs, perm, kept, remap);
    // 6) gather + tanh gating into hbuf
    gather_gate<<<dim3(k, BB), HH, 0, stream>>>(hgcn, scs, perm, hbuf, n, k);
    // 7) readout (max | mean), accumulated across stages
    readout_acc<<<BB, HH, 0, stream>>>(hbuf, k, rbuf);
    // 8) remap/zero edges for next stage (in scratch copies only)
    edge_filter<<<(totalE + 255) / 256, 256, 0, stream>>>(esrc, edst, ew, kept, remap, n, totalE);
    hin = hbuf;
    n = k;
  }
  mlp_head<<<BB, 128, 0, stream>>>(rbuf,
                                   (const float*)d_in[15], (const float*)d_in[16],
                                   (const float*)d_in[17], (const float*)d_in[18],
                                   (const float*)d_in[19], (const float*)d_in[20],
                                   (float*)d_out);
}
